// PhiMoESparseMoeBlock_14413910245654
// MI455X (gfx1250) — matrix operand order repaired and checked by reading, not compiled, not run on hardware
//
#include <hip/hip_runtime.h>
#include <hip/hip_bf16.h>

// ---------------------------------------------------------------------------
// PhiMoE sparse MoE block for MI455X (gfx1250, wave32, WMMA).
// T=1024 tokens, H=1024, I=2048, E=8 experts, top-2 routing.
//
// Pipeline:
//   1) memset out + expert counters
//   2) gate kernel: wave-per-token logits, sparse-mixer top-2, compacted
//      per-expert token lists (atomics)
//   3) weight pre-convert: fp32 -> bf16 cache in d_ws (101 MB, L2-resident)
//   4) FFN kernel: 32-token x expert x I-split tiles, bf16 WMMA (f32 acc),
//      SiLU(h1)*h3 chunk in LDS, 16 register-resident y tiles per wave,
//      gate-scaled atomic scatter into out (atomics make the I-split free).
// ---------------------------------------------------------------------------

#define T_TOK 1024
#define HDIM  1024
#define IDIM  2048
#define NEXP  8
#define TILE_M 32
#define CSPLIT 4                       // I-chunk range split across blocks
#define NCHUNK (IDIM / 128)            // 16 chunks of 128 intermediate chans
#define XS (HDIM + 8)   // padded bf16 row stride, x tile in LDS (no bank conflicts)
#define AS (128 + 8)    // padded bf16 row stride, act chunk in LDS

#define NELEM ((size_t)NEXP * IDIM * HDIM)   // 16.78M elements per weight tensor

// d_ws layout
#define WS_CNT  0
#define WS_TIDX 256
#define WS_TW   (256 + (size_t)NEXP * T_TOK * sizeof(int))
#define WS_WB   131072                        // bf16 weight cache base

typedef __attribute__((ext_vector_type(16))) __bf16        v16bf;
typedef __attribute__((ext_vector_type(8)))  float         v8f;
typedef __attribute__((ext_vector_type(8)))  unsigned int  v8u;

__device__ __forceinline__ unsigned int packbf(float a, float b) {
    unsigned short lo = __builtin_bit_cast(unsigned short, (__bf16)a);
    unsigned short hi = __builtin_bit_cast(unsigned short, (__bf16)b);
    return (unsigned int)lo | ((unsigned int)hi << 16);
}

// A fragment (16x32 bf16, MxK) from row-major bf16 LDS (contiguous K layout:
// lane m = lane&15, upper half-wave carries K 16..31, 16 consecutive K per lane).
__device__ __forceinline__ v16bf a_frag_lds(const __bf16* base, int stride,
                                            int k0, int lane) {
    const int m  = lane & 15;
    const int ko = (lane >> 4) << 4;   // 0 or 16
    const unsigned int* p0 =
        (const unsigned int*)(base + (size_t)m * stride + k0 + ko);
    const unsigned int* p1 =
        (const unsigned int*)(base + (size_t)m * stride + k0 + 8 + ko);
    v8u u;
#pragma unroll
    for (int j = 0; j < 4; ++j) { u[j] = p0[j]; u[j + 4] = p1[j]; }
    return __builtin_bit_cast(v16bf, u);
}

// B fragment (32x16, KxN): B[k][n] = W[n0+n][k0+k], W row-major bf16.
// 16 consecutive bf16 per lane -> one 32-byte vector load (2x b128).
__device__ __forceinline__ v16bf b_frag_bf16(const __bf16* __restrict__ W,
                                             int rowStride, int n0, int k0,
                                             int lane) {
    const int nn = lane & 15;
    const int kb = (lane >> 4) << 4;   // 0 or 16
    return *reinterpret_cast<const v16bf*>(
        W + (size_t)(n0 + nn) * rowStride + k0 + kb);
}

// ---------------------------------------------------------------------------
// Weight pre-conversion: fp32 -> bf16, 8 elements per thread, fully coalesced.
// ---------------------------------------------------------------------------
__global__ __launch_bounds__(256) void cvt_weights(const float* __restrict__ src,
                                                   __bf16* __restrict__ dst) {
    const size_t i = ((size_t)blockIdx.x * 256 + threadIdx.x) * 8;
    const float4* s = reinterpret_cast<const float4*>(src + i);
    const float4 a = s[0], b = s[1];
    uint4 o;
    o.x = packbf(a.x, a.y); o.y = packbf(a.z, a.w);
    o.z = packbf(b.x, b.y); o.w = packbf(b.z, b.w);
    *reinterpret_cast<uint4*>(dst + i) = o;
}

// ---------------------------------------------------------------------------
// Gating: one wave32 per token. logits = x @ gate_w^T, sparse-mixer top-2,
// atomic append into per-expert compacted token lists.
// ---------------------------------------------------------------------------
__global__ __launch_bounds__(256) void moe_gate(
    const float* __restrict__ x, const float* __restrict__ gw,
    int* __restrict__ cnt, int* __restrict__ tIdx, float* __restrict__ tW) {
    __shared__ float g[NEXP * HDIM];   // 32 KB
    const int tid = threadIdx.x;
    for (int i = tid; i < NEXP * HDIM; i += 256) g[i] = gw[i];
    __syncthreads();

    const int lane = tid & 31;
    const int wv   = tid >> 5;
    const int t    = blockIdx.x * 8 + wv;

    float acc[NEXP];
#pragma unroll
    for (int e = 0; e < NEXP; ++e) acc[e] = 0.f;
    const float* xr = x + (size_t)t * HDIM;
    for (int it = 0; it < HDIM / 32; ++it) {
        const int h = lane + it * 32;
        const float xv = xr[h];
#pragma unroll
        for (int e = 0; e < NEXP; ++e) acc[e] += xv * g[e * HDIM + h];
    }
#pragma unroll
    for (int e = 0; e < NEXP; ++e)
#pragma unroll
        for (int off = 16; off > 0; off >>= 1)
            acc[e] += __shfl_xor(acc[e], off, 32);

    if (lane == 0) {
        float s[NEXP];
#pragma unroll
        for (int e = 0; e < NEXP; ++e) s[e] = acc[e];
        int i1 = 0; float t1 = s[0];
        for (int e = 1; e < NEXP; ++e) if (s[e] > t1) { t1 = s[e]; i1 = e; }
        int i2 = (i1 == 0) ? 1 : 0; float t2 = s[i2];
        for (int e = 0; e < NEXP; ++e)
            if (e != i1 && s[e] > t2) { t2 = s[e]; i2 = e; }

        const float thr = 0.02f;   // 2 * JITTER_EPS
        float d1 = 0.f;
        for (int e = 0; e < NEXP; ++e) {
            float f = fmaxf(fabsf(s[e]), t1);
            if (!(((t1 - s[e]) / f) > thr)) d1 += __expf(s[e] - t1);
        }
        const float mult1 = 1.f / d1;           // top1 is never masked
        float d2 = 0.f;
        for (int e = 0; e < NEXP; ++e) {
            if (e == i1) continue;              // s2[i1] = -inf -> exp = 0
            float f = fmaxf(fabsf(s[e]), t2);
            if (!(((t2 - s[e]) / f) > thr)) d2 += __expf(s[e] - t2);
        }
        const float mult2 = 1.f / d2;

        int p = atomicAdd(&cnt[i1], 1);
        tIdx[i1 * T_TOK + p] = t; tW[i1 * T_TOK + p] = mult1;
        p = atomicAdd(&cnt[i2], 1);
        tIdx[i2 * T_TOK + p] = t; tW[i2 * T_TOK + p] = mult2;
    }
}

// ---------------------------------------------------------------------------
// Expert FFN: block = one (expert, 32-token tile, I-chunk split). 8 waves.
// Per 128-wide I-chunk: wave w computes two 16x16 (h1,h3) tiles (K=H, bf16
// WMMA, B fragments shared across both m-tiles), SiLU*h3 -> bf16 act chunk
// in LDS, then every wave accumulates its 16 register-resident 16x16 y tiles
// over the chunk (K=128, b2 shared across m-tiles). Finally scale by gate
// weight and atomically scatter into out.
// ---------------------------------------------------------------------------
__global__ __launch_bounds__(256) void moe_ffn(
    const float* __restrict__ x,
    const __bf16* __restrict__ w1b, const __bf16* __restrict__ w2b,
    const __bf16* __restrict__ w3b,
    const int* __restrict__ cnt,  const int* __restrict__ tIdx,
    const float* __restrict__ tW, float* __restrict__ out) {
    __shared__ __bf16 xs[TILE_M * XS];      // 66 KB  x tile (bf16)
    __shared__ __bf16 acts[TILE_M * AS];    // 8.7 KB act chunk (bf16)
    __shared__ int    tok[TILE_M];
    __shared__ float  gws[TILE_M];

    const int e  = blockIdx.y;
    const int n  = cnt[e];
    const int t0 = blockIdx.x * TILE_M;
    if (t0 >= n) return;                    // block-uniform early exit

    const int tid  = threadIdx.x;
    const int lane = tid & 31;
    const int wv   = tid >> 5;
    const int nn   = lane & 15;
    const int mo   = (lane >> 4) << 3;      // C-tile row offset: 0 or 8

    if (tid < TILE_M) {
        const int idx = t0 + tid;
        if (idx < n) { tok[tid] = tIdx[e * T_TOK + idx]; gws[tid] = tW[e * T_TOK + idx]; }
        else         { tok[tid] = 0;                     gws[tid] = 0.f; }
    }
    __syncthreads();

    // Stage gathered x rows into LDS as bf16 (float4 loads, packed b32 stores).
    for (int it = tid; it < TILE_M * (HDIM / 4); it += 256) {
        const int row = it >> 8;            // HDIM/4 = 256 float4 per row
        const int c4  = it & 255;
        const float4 v = reinterpret_cast<const float4*>(
            x + (size_t)tok[row] * HDIM)[c4];
        unsigned int* d = (unsigned int*)(xs + row * XS + c4 * 4);
        d[0] = packbf(v.x, v.y);
        d[1] = packbf(v.z, v.w);
    }
    __syncthreads();

    const __bf16* W1 = w1b + (size_t)e * IDIM * HDIM;
    const __bf16* W3 = w3b + (size_t)e * IDIM * HDIM;
    const __bf16* W2 = w2b + (size_t)e * HDIM * IDIM;

    v8f y[16];                              // [h-tile j][m-tile] accumulators
#pragma unroll
    for (int j = 0; j < 16; ++j) y[j] = (v8f){0.f,0.f,0.f,0.f,0.f,0.f,0.f,0.f};

    const int cBeg = blockIdx.z * (NCHUNK / CSPLIT);
    const int cEnd = cBeg + (NCHUNK / CSPLIT);

    for (int c = cBeg; c < cEnd; ++c) {
        const int ib = c * 128 + wv * 16;   // this wave's intermediate tile
        if (c + 1 < cEnd) {                 // warm caches for next chunk's rows
            const int ibn = (c + 1) * 128 + wv * 16;
            __builtin_prefetch(W1 + (size_t)(ibn + nn) * HDIM, 0, 3);
            __builtin_prefetch(W3 + (size_t)(ibn + nn) * HDIM, 0, 3);
        }
        v8f c1a = (v8f){0.f,0.f,0.f,0.f,0.f,0.f,0.f,0.f};
        v8f c1b = c1a, c3a = c1a, c3b = c1a;
#pragma unroll 2
        for (int ks = 0; ks < HDIM / 32; ++ks) {
            const int k0 = ks * 32;
            v16bf a0 = a_frag_lds(xs,               XS, k0, lane);
            v16bf a1 = a_frag_lds(xs + 16 * XS,     XS, k0, lane);
            v16bf b1 = b_frag_bf16(W1, HDIM, ib, k0, lane);
            v16bf b3 = b_frag_bf16(W3, HDIM, ib, k0, lane);
            c1a = __builtin_amdgcn_wmma_f32_16x16x32_bf16(false, a0, false, b1,
                                                          (short)0, c1a, false, false);
            c1b = __builtin_amdgcn_wmma_f32_16x16x32_bf16(false, a1, false, b1,
                                                          (short)0, c1b, false, false);
            c3a = __builtin_amdgcn_wmma_f32_16x16x32_bf16(false, a0, false, b3,
                                                          (short)0, c3a, false, false);
            c3b = __builtin_amdgcn_wmma_f32_16x16x32_bf16(false, a1, false, b3,
                                                          (short)0, c3b, false, false);
        }
        __syncthreads();                    // prior chunk's act fully consumed
#pragma unroll
        for (int r = 0; r < 8; ++r) {       // C layout: row = r + mo, col = nn
            const float s0 = __builtin_amdgcn_rcpf(1.f + __expf(-c1a[r]));
            acts[(r + mo) * AS + wv * 16 + nn]      = (__bf16)(c1a[r] * s0 * c3a[r]);
            const float s1 = __builtin_amdgcn_rcpf(1.f + __expf(-c1b[r]));
            acts[(16 + r + mo) * AS + wv * 16 + nn] = (__bf16)(c1b[r] * s1 * c3b[r]);
        }
        __syncthreads();                    // act chunk visible to all waves
#pragma unroll
        for (int ks = 0; ks < 4; ++ks) {    // K = 128 per chunk
            v16bf a20 = a_frag_lds(acts,           AS, ks * 32, lane);
            v16bf a21 = a_frag_lds(acts + 16 * AS, AS, ks * 32, lane);
#pragma unroll
            for (int j = 0; j < 8; ++j) {   // 8 H-tiles per wave -> 64 total
                const int h0 = wv * 16 + j * 128;
                v16bf b2 = b_frag_bf16(W2, IDIM, h0, c * 128 + ks * 32, lane);
                y[2 * j]     = __builtin_amdgcn_wmma_f32_16x16x32_bf16(
                                   false, a20, false, b2, (short)0, y[2 * j],     false, false);
                y[2 * j + 1] = __builtin_amdgcn_wmma_f32_16x16x32_bf16(
                                   false, a21, false, b2, (short)0, y[2 * j + 1], false, false);
            }
        }
    }

    // Scale by gate weight and scatter-add (2 experts x CSPLIT partials/token).
#pragma unroll
    for (int j = 0; j < 8; ++j) {
        const int h0 = wv * 16 + j * 128;
#pragma unroll
        for (int r = 0; r < 8; ++r) {
            const int m0 = r + mo;
            atomicAdd(out + (size_t)tok[m0] * HDIM + h0 + nn, y[2 * j][r] * gws[m0]);
            const int m1 = 16 + r + mo;
            atomicAdd(out + (size_t)tok[m1] * HDIM + h0 + nn, y[2 * j + 1][r] * gws[m1]);
        }
    }
}

// ---------------------------------------------------------------------------
extern "C" void kernel_launch(void* const* d_in, const int* in_sizes, int n_in,
                              void* d_out, int out_size, void* d_ws, size_t ws_size,
                              hipStream_t stream) {
    const float* x  = (const float*)d_in[0];
    const float* gw = (const float*)d_in[1];
    const float* w1 = (const float*)d_in[2];
    const float* w2 = (const float*)d_in[3];
    const float* w3 = (const float*)d_in[4];
    float* out = (float*)d_out;

    int*    cnt  = (int*)((char*)d_ws + WS_CNT);
    int*    tIdx = (int*)((char*)d_ws + WS_TIDX);
    float*  tW   = (float*)((char*)d_ws + WS_TW);
    __bf16* w1b  = (__bf16*)((char*)d_ws + WS_WB);
    __bf16* w3b  = w1b + NELEM;
    __bf16* w2b  = w3b + NELEM;

    hipMemsetAsync(d_out, 0, (size_t)out_size * sizeof(float), stream);
    hipMemsetAsync(d_ws, 0, 256, stream);

    moe_gate<<<T_TOK / 8, 256, 0, stream>>>(x, gw, cnt, tIdx, tW);

    const int cvtBlocks = (int)(NELEM / (256 * 8));   // 8 elems per thread
    cvt_weights<<<cvtBlocks, 256, 0, stream>>>(w1, w1b);
    cvt_weights<<<cvtBlocks, 256, 0, stream>>>(w3, w3b);
    cvt_weights<<<cvtBlocks, 256, 0, stream>>>(w2, w2b);

    moe_ffn<<<dim3(T_TOK / TILE_M, NEXP, CSPLIT), 256, 0, stream>>>(
        x, w1b, w2b, w3b, cnt, tIdx, tW, out);
}
